// Encoder_49151605735821
// MI455X (gfx1250) — matrix operand order repaired and checked by reading, not compiled, not run on hardware
//
#include <hip/hip_runtime.h>
#include <hip/hip_bf16.h>
#include <math.h>

// ---------------- types ----------------
typedef __bf16 bf16_t;
typedef __attribute__((ext_vector_type(16))) __bf16 v16bf;
typedef __attribute__((ext_vector_type(8)))  __bf16 v8bf;
typedef __attribute__((ext_vector_type(8)))  float  v8f;
typedef __attribute__((ext_vector_type(4)))  unsigned int v4u;
typedef __attribute__((ext_vector_type(8)))  int v8i;
typedef __attribute__((ext_vector_type(4)))  int v4i;

// ---------------- constants ----------------
#define BATCH 64
#define SEQ   2048
#define H1DIM 64
#define H2DIM 128
#define DICT  1024
#define LN_EPS 1e-5f

#if defined(__has_builtin)
#if __has_builtin(__builtin_amdgcn_tensor_load_to_lds) && __has_builtin(__builtin_amdgcn_s_wait_tensorcnt)
#define HAVE_TDM 1
#endif
#if __has_builtin(__builtin_amdgcn_tanhf)
#define HAVE_TANH 1
#endif
#endif

__device__ __forceinline__ float tanh_fast(float u) {
#ifdef HAVE_TANH
    return __builtin_amdgcn_tanhf(u);          // native v_tanh_f32
#else
    // branch-free: tanh(|u|) = 1 - 2e/(1+e), e = exp(-2|u|) in (0,1]
    float e  = __expf(-2.0f * fabsf(u));
    float th = 1.0f - 2.0f * e * __builtin_amdgcn_rcpf(1.0f + e);
    return copysignf(th, u);
#endif
}

__device__ __forceinline__ float gelu_fast(float x) {
    // tanh-form GeLU; error << bf16 quantization of the stored result.
    float u = 0.7978845608028654f * (x + 0.044715f * x * x * x);
    return 0.5f * x * (1.0f + tanh_fast(u));
}

__device__ __forceinline__ float bfbits_to_f32(unsigned int lo16) {
    union { unsigned int u; float f; } c;
    c.u = lo16 << 16;
    return c.f;
}

#ifdef HAVE_TDM
// 2D tensor_load_to_lds: rows x rowElems of 2-byte data, optional LDS padding
// (padAmountCode+1 DWORDs inserted every 2<<padIntervalCode DWORDs).
// This toolchain's builtin takes 6 args: (g0, g1, g2, g3, g_extra, cpol).
__device__ __forceinline__ void tdm_load_2d_bf16(unsigned lds_byte_off,
                                                 const void* gaddr,
                                                 unsigned rowElems, unsigned rows,
                                                 unsigned rowStrideElems,
                                                 unsigned padIntervalCode,
                                                 unsigned padAmountCode,
                                                 unsigned padEnable) {
    unsigned long long ga = (unsigned long long)(size_t)gaddr;
    v4u g0;
    g0[0] = 1u;                                            // count=1, user desc
    g0[1] = lds_byte_off;                                  // lds_addr
    g0[2] = (unsigned)(ga & 0xFFFFFFFFu);                  // global_addr[31:0]
    g0[3] = (unsigned)((ga >> 32) & 0x01FFFFFFu) | (2u << 30); // addr[56:32] | type=2
    v8i g1;
    g1[0] = (int)((1u << 16) |                             // data_size = 2 bytes
                  (padEnable << 20) |
                  (padIntervalCode << 22) |
                  (padAmountCode << 25));
    g1[1] = (int)((rowElems & 0xFFFFu) << 16);             // tensor_dim0[15:0]
    g1[2] = (int)(((rowElems >> 16) & 0xFFFFu) |           // tensor_dim0[31:16]
                  ((rows & 0xFFFFu) << 16));               // tensor_dim1[15:0]
    g1[3] = (int)(((rows >> 16) & 0xFFFFu) |               // tensor_dim1[31:16]
                  ((rowElems & 0xFFFFu) << 16));           // tile_dim0
    g1[4] = (int)(rows & 0xFFFFu);                         // tile_dim1 (tile_dim2=0)
    g1[5] = (int)rowStrideElems;                           // tensor_dim0_stride[31:0]
    g1[6] = 0;
    g1[7] = 0;
    v4i gz4 = {0, 0, 0, 0};
    v8i gz8 = {0, 0, 0, 0, 0, 0, 0, 0};
    __builtin_amdgcn_tensor_load_to_lds(g0, g1, gz4, gz4, gz8, 0);
}
#endif

// ============================================================
// Kernel 0: w2 [64x128] -> w2t [128][64] bf16 (transposed),
//           w3 [128x1024] -> w3t [1024][128] bf16 (transposed)
// ============================================================
__global__ void cvt_weights_kernel(const float* __restrict__ w2,
                                   const float* __restrict__ w3,
                                   bf16_t* __restrict__ w2t,
                                   bf16_t* __restrict__ w3t) {
    int i = blockIdx.x * blockDim.x + threadIdx.x;   // 131072 threads
    if (i < H1DIM * H2DIM) {
        int k = i / H2DIM, n = i % H2DIM;
        w2t[n * H1DIM + k] = (bf16_t)w2[i];
    }
    int k3 = i / DICT, n3 = i % DICT;
    w3t[(size_t)n3 * H2DIM + k3] = (bf16_t)w3[i];
}

// ============================================================
// per-batch mean/rstd, f32 source
// ============================================================
__global__ __launch_bounds__(256)
void stats_f32_kernel(const float* __restrict__ src, int perBatch,
                      float* __restrict__ stats) {
    int b = blockIdx.x;
    const float4* p = (const float4*)(src + (size_t)b * perBatch);
    int n4 = perBatch >> 2;
    float s = 0.f, s2 = 0.f;
    for (int i = threadIdx.x; i < n4; i += blockDim.x) {
        float4 v = p[i];
        s  += v.x + v.y + v.z + v.w;
        s2 += v.x*v.x + v.y*v.y + v.z*v.z + v.w*v.w;
    }
    __shared__ float r0[256], r1[256];
    r0[threadIdx.x] = s; r1[threadIdx.x] = s2;
    __syncthreads();
    for (int off = 128; off > 0; off >>= 1) {
        if ((int)threadIdx.x < off) {
            r0[threadIdx.x] += r0[threadIdx.x + off];
            r1[threadIdx.x] += r1[threadIdx.x + off];
        }
        __syncthreads();
    }
    if (threadIdx.x == 0) {
        float inv = 1.0f / (float)perBatch;
        float mu  = r0[0] * inv;
        float var = r1[0] * inv - mu * mu;
        stats[2*b]   = mu;
        stats[2*b+1] = rsqrtf(var + LN_EPS);
    }
}

// ============================================================
// per-batch mean/rstd, bf16 source
// ============================================================
__global__ __launch_bounds__(256)
void stats_bf16_kernel(const bf16_t* __restrict__ src, int perBatch,
                       float* __restrict__ stats) {
    int b = blockIdx.x;
    const uint4* p = (const uint4*)(src + (size_t)b * perBatch);
    int n8 = perBatch >> 3;
    float s = 0.f, s2 = 0.f;
    for (int i = threadIdx.x; i < n8; i += blockDim.x) {
        uint4 v = p[i];
        unsigned int ws[4] = { v.x, v.y, v.z, v.w };
        #pragma unroll
        for (int j = 0; j < 4; j++) {
            float a = bfbits_to_f32(ws[j] & 0xFFFFu);
            float c = bfbits_to_f32(ws[j] >> 16);
            s  += a + c;
            s2 += a*a + c*c;
        }
    }
    __shared__ float r0[256], r1[256];
    r0[threadIdx.x] = s; r1[threadIdx.x] = s2;
    __syncthreads();
    for (int off = 128; off > 0; off >>= 1) {
        if ((int)threadIdx.x < off) {
            r0[threadIdx.x] += r0[threadIdx.x + off];
            r1[threadIdx.x] += r1[threadIdx.x + off];
        }
        __syncthreads();
    }
    if (threadIdx.x == 0) {
        float inv = 1.0f / (float)perBatch;
        float mu  = r0[0] * inv;
        float var = r1[0] * inv - mu * mu;
        stats[2*b]   = mu;
        stats[2*b+1] = rsqrtf(var + LN_EPS);
    }
}

// ============================================================
// LN1 + (2 -> 64) matvec + bias + GeLU -> h1 bf16
// ============================================================
__global__ __launch_bounds__(256)
void l1_kernel(const float* __restrict__ in,   // [B, 2, S]
               const float* __restrict__ ln1w, // [S, 2]
               const float* __restrict__ ln1b, // [S, 2]
               const float* __restrict__ w1,   // [2, 64]
               const float* __restrict__ b1,   // [64]
               const float* __restrict__ stats1,
               bf16_t* __restrict__ h1) {      // [B, S, 64]
    int b = blockIdx.y;
    int s = blockIdx.x * blockDim.x + threadIdx.x;

    __shared__ float sw[128], sb[64];
    if (threadIdx.x < 128) sw[threadIdx.x] = w1[threadIdx.x];
    if (threadIdx.x < 64)  sb[threadIdx.x] = b1[threadIdx.x];
    __syncthreads();

    float mu = stats1[2*b], rs = stats1[2*b+1];
    float x0 = in[(size_t)b * 2 * SEQ + s];
    float x1 = in[(size_t)b * 2 * SEQ + SEQ + s];
    float y0 = (x0 - mu) * rs * ln1w[2*s]   + ln1b[2*s];
    float y1 = (x1 - mu) * rs * ln1w[2*s+1] + ln1b[2*s+1];

    bf16_t* out = h1 + ((size_t)b * SEQ + s) * H1DIM;
    #pragma unroll
    for (int j0 = 0; j0 < H1DIM; j0 += 8) {
        v8bf pk;
        #pragma unroll
        for (int e = 0; e < 8; e++) {
            int j = j0 + e;
            float v = fmaf(y0, sw[j], fmaf(y1, sw[64 + j], sb[j]));
            pk[e] = (bf16_t)gelu_fast(v);
        }
        *(v8bf*)(out + j0) = pk;
    }
}

// ============================================================
// LN2 + GEMM (K=64 -> N=128) + GeLU via bf16 WMMA.
// Block = 8 waves = 128 rows. A tile staged raw by TDM into
// padded LDS (stride 72 bf16), normalized in place, fragments
// loaded as ds b128; B fragments are contiguous v16bf from w2t.
// ============================================================
__global__ __launch_bounds__(256)
void l2_kernel(const bf16_t* __restrict__ h1,   // [B, S, 64]
               const float* __restrict__ ln2w,  // [S, 64]
               const float* __restrict__ ln2b,  // [S, 64]
               const bf16_t* __restrict__ w2t,  // [128][64] (N-major)
               const float* __restrict__ b2,    // [128]
               const float* __restrict__ stats2,
               bf16_t* __restrict__ h2) {       // [B, S, 128]
    int b      = blockIdx.x;
    int stile  = blockIdx.y;
    int w      = threadIdx.x >> 5;
    int lane   = threadIdx.x & 31;
    int sBase0 = stile * 128;          // block rows
    int sBase  = sBase0 + w * 16;      // wave rows

    float mu = stats2[2*b], rs = stats2[2*b+1];

    __shared__ bf16_t aT[128][72];     // raw tile, TDM-padded stride

    const bf16_t* gtile = h1 + ((size_t)b * SEQ + sBase0) * H1DIM;
#ifdef HAVE_TDM
    if (threadIdx.x == 0) {
        unsigned lds_off = (unsigned)(size_t)(void*)&aT[0][0];
        // row = 64 bf16 = 32 DWORDs -> interval code 4; pad 4 DWORDs -> code 3
        tdm_load_2d_bf16(lds_off, gtile, H1DIM, 128, H1DIM, 4, 3, 1);
        __builtin_amdgcn_s_wait_tensorcnt(0);
    }
#else
    for (int it = 0; it < 4; it++) {   // 128 rows * 8 chunks / 256 thr
        int cid = it * 256 + threadIdx.x;
        int r = cid >> 3, c8 = (cid & 7) * 8;
        *(uint4*)&aT[r][c8] = *(const uint4*)&gtile[r * H1DIM + c8];
    }
#endif
    __syncthreads();

    // in-place LN2 on this wave's 16 rows
    #pragma unroll
    for (int it = 0; it < 4; it++) {   // 16 rows * 8 chunks / 32 lanes
        int cid = it * 32 + lane;
        int r = cid >> 3, c8 = (cid & 7) * 8;
        int srow = sBase + r;
        v8bf vv = *(v8bf*)&aT[w * 16 + r][c8];
        const float* lw = &ln2w[srow * H1DIM + c8];
        const float* lb = &ln2b[srow * H1DIM + c8];
        #pragma unroll
        for (int e = 0; e < 8; e++) {
            float f = (float)vv[e];
            vv[e] = (bf16_t)((f - mu) * rs * lw[e] + lb[e]);
        }
        *(v8bf*)&aT[w * 16 + r][c8] = vv;
    }
    __syncthreads();

    int m  = lane & 15;
    int hi = lane >> 4;
    int n  = lane & 15;

    // A fragments: one contiguous 16-elem run (k = 16*hi + s) per 32-K chunk
    v16bf a0, a1;
    {
        v8bf p0 = *(v8bf*)&aT[w * 16 + m][hi * 16];
        v8bf p1 = *(v8bf*)&aT[w * 16 + m][hi * 16 + 8];
        v8bf p2 = *(v8bf*)&aT[w * 16 + m][32 + hi * 16];
        v8bf p3 = *(v8bf*)&aT[w * 16 + m][32 + hi * 16 + 8];
        a0 = __builtin_shufflevector(p0, p1, 0,1,2,3,4,5,6,7,8,9,10,11,12,13,14,15);
        a1 = __builtin_shufflevector(p2, p3, 0,1,2,3,4,5,6,7,8,9,10,11,12,13,14,15);
    }

    bf16_t* orow = h2 + ((size_t)b * SEQ + sBase) * H2DIM;
    for (int nt = 0; nt < 8; nt++) {
        int ncol = nt * 16 + n;
        float bias = b2[ncol];
        v8f c;
        #pragma unroll
        for (int r = 0; r < 8; r++) c[r] = bias;

        const bf16_t* wrow = &w2t[(size_t)ncol * H1DIM];
        v16bf b0 = *(const v16bf*)(wrow + hi * 16);
        v16bf b1v = *(const v16bf*)(wrow + 32 + hi * 16);

        c = __builtin_amdgcn_wmma_f32_16x16x32_bf16(false, a0, false, b0,
                                                    (short)0, c, false, false);
        c = __builtin_amdgcn_wmma_f32_16x16x32_bf16(false, a1, false, b1v,
                                                    (short)0, c, false, false);

        #pragma unroll
        for (int r = 0; r < 8; r++) {
            int row = r + hi * 8;
            orow[(size_t)row * H2DIM + ncol] = (bf16_t)gelu_fast(c[r]);
        }
    }
}

// ============================================================
// LN3 + GEMM (K=128 -> N=1024) + bias + argmax (WMMA).
// Same staging scheme; 64 N-tiles with running per-lane argmax,
// then 16-lane shfl_xor (max, first-index) reduction.
// ============================================================
__global__ __launch_bounds__(256)
void l3_kernel(const bf16_t* __restrict__ h2,   // [B, S, 128]
               const float* __restrict__ ln3w,  // [S, 128]
               const float* __restrict__ ln3b,  // [S, 128]
               const bf16_t* __restrict__ w3t,  // [1024][128] (N-major)
               const float* __restrict__ b3,    // [1024]
               const float* __restrict__ stats3,
               int* __restrict__ out) {         // [B, S]
    int b      = blockIdx.x;
    int stile  = blockIdx.y;
    int w      = threadIdx.x >> 5;
    int lane   = threadIdx.x & 31;
    int sBase0 = stile * 128;
    int sBase  = sBase0 + w * 16;

    float mu = stats3[2*b], rs = stats3[2*b+1];

    __shared__ bf16_t aT[128][136];    // raw tile, TDM-padded stride

    const bf16_t* gtile = h2 + ((size_t)b * SEQ + sBase0) * H2DIM;
#ifdef HAVE_TDM
    if (threadIdx.x == 0) {
        unsigned lds_off = (unsigned)(size_t)(void*)&aT[0][0];
        // row = 128 bf16 = 64 DWORDs -> interval code 5; pad 4 DWORDs -> code 3
        tdm_load_2d_bf16(lds_off, gtile, H2DIM, 128, H2DIM, 5, 3, 1);
        __builtin_amdgcn_s_wait_tensorcnt(0);
    }
#else
    for (int it = 0; it < 8; it++) {   // 128 rows * 16 chunks / 256 thr
        int cid = it * 256 + threadIdx.x;
        int r = cid >> 4, c8 = (cid & 15) * 8;
        *(uint4*)&aT[r][c8] = *(const uint4*)&gtile[r * H2DIM + c8];
    }
#endif
    __syncthreads();

    // in-place LN3 on this wave's 16 rows
    #pragma unroll
    for (int it = 0; it < 8; it++) {   // 16 rows * 16 chunks / 32 lanes
        int cid = it * 32 + lane;
        int r = cid >> 4, c8 = (cid & 15) * 8;
        int srow = sBase + r;
        v8bf vv = *(v8bf*)&aT[w * 16 + r][c8];
        const float* lw = &ln3w[srow * H2DIM + c8];
        const float* lb = &ln3b[srow * H2DIM + c8];
        #pragma unroll
        for (int e = 0; e < 8; e++) {
            float f = (float)vv[e];
            vv[e] = (bf16_t)((f - mu) * rs * lw[e] + lb[e]);
        }
        *(v8bf*)&aT[w * 16 + r][c8] = vv;
    }
    __syncthreads();

    int m  = lane & 15;
    int hi = lane >> 4;
    int n  = lane & 15;

    v16bf a[4];
    #pragma unroll
    for (int ch = 0; ch < 4; ch++) {
        v8bf plo = *(v8bf*)&aT[w * 16 + m][ch * 32 + hi * 16];
        v8bf phi = *(v8bf*)&aT[w * 16 + m][ch * 32 + hi * 16 + 8];
        a[ch] = __builtin_shufflevector(plo, phi, 0,1,2,3,4,5,6,7,8,9,10,11,12,13,14,15);
    }

    float bestv[8];
    int   besti[8];
    #pragma unroll
    for (int r = 0; r < 8; r++) { bestv[r] = -INFINITY; besti[r] = 0; }

    for (int nt = 0; nt < 64; nt++) {
        int ncol = nt * 16 + n;
        float bias = b3[ncol];
        v8f c;
        #pragma unroll
        for (int r = 0; r < 8; r++) c[r] = bias;

        if (nt + 1 < 64) {  // uniform branch
            __builtin_prefetch(&w3t[(size_t)(ncol + 16) * H2DIM], 0, 0);
        }

        const bf16_t* wrow = &w3t[(size_t)ncol * H2DIM];
        #pragma unroll
        for (int ch = 0; ch < 4; ch++) {
            v16bf bv = *(const v16bf*)(wrow + ch * 32 + hi * 16);
            c = __builtin_amdgcn_wmma_f32_16x16x32_bf16(false, a[ch], false, bv,
                                                        (short)0, c, false, false);
        }

        #pragma unroll
        for (int r = 0; r < 8; r++) {
            if (c[r] > bestv[r]) { bestv[r] = c[r]; besti[r] = ncol; }
        }
    }

    #pragma unroll
    for (int mask = 8; mask >= 1; mask >>= 1) {
        #pragma unroll
        for (int r = 0; r < 8; r++) {
            float ov = __shfl_xor(bestv[r], mask, 32);
            int   oi = __shfl_xor(besti[r], mask, 32);
            if (ov > bestv[r] || (ov == bestv[r] && oi < besti[r])) {
                bestv[r] = ov; besti[r] = oi;
            }
        }
    }

    if ((lane & 15) == 0) {
        int* orow = out + (size_t)b * SEQ + sBase + hi * 8;
        #pragma unroll
        for (int r = 0; r < 8; r++) orow[r] = besti[r];
    }
}

// ============================================================
// host launcher
// ============================================================
extern "C" void kernel_launch(void* const* d_in, const int* in_sizes, int n_in,
                              void* d_out, int out_size, void* d_ws, size_t ws_size,
                              hipStream_t stream) {
    const float* in    = (const float*)d_in[0];
    const float* ln1w  = (const float*)d_in[1];
    const float* ln1b  = (const float*)d_in[2];
    const float* w1    = (const float*)d_in[3];
    const float* b1    = (const float*)d_in[4];
    const float* ln2w  = (const float*)d_in[5];
    const float* ln2b  = (const float*)d_in[6];
    const float* w2    = (const float*)d_in[7];
    const float* b2    = (const float*)d_in[8];
    const float* ln3w  = (const float*)d_in[9];
    const float* ln3b  = (const float*)d_in[10];
    const float* w3    = (const float*)d_in[11];
    const float* b3    = (const float*)d_in[12];
    int* out = (int*)d_out;

    char* ws = (char*)d_ws;
    float*  stats1 = (float*)(ws + 0);
    float*  stats2 = (float*)(ws + 512);
    float*  stats3 = (float*)(ws + 1024);
    bf16_t* w2t    = (bf16_t*)(ws + 2048);        // [128][64]  bf16, 16 KB
    bf16_t* w3t    = (bf16_t*)(ws + 18432);       // [1024][128] bf16, 256 KB
    bf16_t* h1     = (bf16_t*)(ws + 280576);      // 16 MB
    bf16_t* h2     = (bf16_t*)(ws + 280576 + (size_t)BATCH*SEQ*H1DIM*2); // 32 MB

    (void)in_sizes; (void)n_in; (void)out_size; (void)ws_size;

    cvt_weights_kernel<<<dim3(512), dim3(256), 0, stream>>>(w2, w3, w2t, w3t);

    stats_f32_kernel<<<dim3(BATCH), dim3(256), 0, stream>>>(in, 2 * SEQ, stats1);

    l1_kernel<<<dim3(SEQ / 256, BATCH), dim3(256), 0, stream>>>(
        in, ln1w, ln1b, w1, b1, stats1, h1);

    stats_bf16_kernel<<<dim3(BATCH), dim3(256), 0, stream>>>(h1, SEQ * H1DIM, stats2);

    l2_kernel<<<dim3(BATCH, SEQ / 128), dim3(256), 0, stream>>>(
        h1, ln2w, ln2b, w2t, b2, stats2, h2);

    stats_bf16_kernel<<<dim3(BATCH), dim3(256), 0, stream>>>(h2, SEQ * H2DIM, stats3);

    l3_kernel<<<dim3(BATCH, SEQ / 128), dim3(256), 0, stream>>>(
        h2, ln3w, ln3b, w3t, b3, stats3, out);
}
